// ViTEncDec_10411000725844
// MI455X (gfx1250) — matrix operand order repaired and checked by reading, not compiled, not run on hardware
//
#include <hip/hip_runtime.h>
#include <hip/hip_bf16.h>
#include <math.h>

// ---------------------------------------------------------------------------
// Types for CDNA5 WMMA (wave32): A/B = v16h (16 halves), C/D = v8f
// ---------------------------------------------------------------------------
typedef __attribute__((ext_vector_type(16))) _Float16 v16h;
typedef __attribute__((ext_vector_type(8)))  _Float16 v8h;
typedef __attribute__((ext_vector_type(8)))  float    v8f;
typedef __attribute__((ext_vector_type(4)))  int      v4i;

#if __has_builtin(__builtin_amdgcn_global_load_async_to_lds_b128) && \
    __has_builtin(__builtin_amdgcn_s_wait_asynccnt)
#define USE_ASYNC_LDS 1
typedef v4i __attribute__((address_space(1)))* gv4i_p;
typedef v4i __attribute__((address_space(3)))* lv4i_p;
#else
#define USE_ASYNC_LDS 0
#endif

// Copy 16 bytes (8 halves) global -> LDS. Async DMA path when available.
__device__ __forceinline__ void copy16_g2l(const _Float16* g, _Float16* l) {
#if USE_ASYNC_LDS
    __builtin_amdgcn_global_load_async_to_lds_b128((gv4i_p)g, (lv4i_p)l, 0, 0);
#else
    *(v8h*)l = *(const v8h*)g;
#endif
}
__device__ __forceinline__ void copy_join() {
#if USE_ASYNC_LDS
    __builtin_amdgcn_s_wait_asynccnt(0);
#endif
}

__device__ __forceinline__ v8f wmma_f16(v16h a, v16h b, v8f c) {
    return __builtin_amdgcn_wmma_f32_16x16x32_f16(
        false, a, false, b, (short)0, c, false, false);
}

// A-fragment (16x32 f16) from LDS tile stored row-major [m][32]
__device__ __forceinline__ v16h load_a32(const _Float16* tile, int m, int kg) {
    const _Float16* q = tile + m * 32;
    v8h lo = *(const v8h*)(q + kg * 16);
    v8h hi = *(const v8h*)(q + kg * 16 + 8);
    v16h r;
#pragma unroll
    for (int i = 0; i < 8; ++i) { r[i] = lo[i]; r[i + 8] = hi[i]; }
    return r;
}

// B-fragment (32x16 f16) from LDS tile stored transposed [n][32]
__device__ __forceinline__ v16h load_b32(const _Float16* tile, int n, int kb) {
    const _Float16* q = tile + n * 32 + kb;
    v8h lo = *(const v8h*)(q);
    v8h hi = *(const v8h*)(q + 8);
    v16h r;
#pragma unroll
    for (int i = 0; i < 8; ++i) { r[i] = lo[i]; r[i + 8] = hi[i]; }
    return r;
}

// ---------------------------------------------------------------------------
// Weight pre-pass:  Wt[n][k] (f16) = W[k][n] (f32).  Coalesced writes.
// ---------------------------------------------------------------------------
__global__ __launch_bounds__(256) void wt_kernel(
    const float* __restrict__ W, _Float16* __restrict__ Wt, int K, int N)
{
    int idx = blockIdx.x * 256 + threadIdx.x;
    if (idx >= K * N) return;
    int k = idx % K, n = idx / K;
    Wt[idx] = (_Float16)W[(size_t)k * N + n];
}

// ---------------------------------------------------------------------------
// WMMA GEMM:  C[M,N] = act(A[M,K] @ W[K,N] + bias) + res
// A: f16 [M][K].  Wt: f16 [N][K] (pre-transposed).  res/bias: f32.
// Output to Cf (f32) or Ch (f16), exactly one non-null.
// 256 thr = 8 waves (4x2), block tile 128x128, wave tile 32x64,
// double-buffered LDS filled by async global->LDS b128 DMA.
// act: 0 none, 1 exact GELU, 2 tanh.  K%32==0, M%128==0, N%4==0.
// ---------------------------------------------------------------------------
__global__ __launch_bounds__(256) void gemm_wmma_kernel(
    const _Float16* __restrict__ A, const _Float16* __restrict__ Wt,
    const float* __restrict__ bias, const float* __restrict__ res,
    float* __restrict__ Cf, _Float16* __restrict__ Ch,
    int M, int N, int K, int act)
{
    __shared__ __align__(16) _Float16 sA[2][128 * 32];   // [m][k]
    __shared__ __align__(16) _Float16 sB[2][128 * 32];   // [n][k]

    const int t    = threadIdx.x;
    const int lane = t & 31;
    const int w    = t >> 5;
    const int wr   = w >> 1;               // 0..3
    const int wc   = w & 1;                // 0..1
    const int bm   = blockIdx.y * 128;
    const int bn   = blockIdx.x * 128;

    // Per-thread staging descriptors: 2 x 16-byte chunks each for A and B.
    const _Float16* aG[2]; const _Float16* bG[2];
    int ldsOff[2]; bool bOk[2];
#pragma unroll
    for (int e = 0; e < 2; ++e) {
        int i   = e * 256 + t;             // 0..511
        int k8  = i & 3;                   // chunk of 8 halves within k-32
        int row = i >> 2;                  // 0..127
        aG[e] = A  + (size_t)(bm + row) * K + k8 * 8;
        bG[e] = Wt + (size_t)(bn + row) * K + k8 * 8;
        bOk[e] = (bn + row) < N;
        ldsOff[e] = row * 32 + k8 * 8;
    }

    v8f acc[2][4] = {};
    const int ktiles = K >> 5;

    // prologue: stage k-tile 0
#pragma unroll
    for (int e = 0; e < 2; ++e) {
        copy16_g2l(aG[e], &sA[0][ldsOff[e]]);
        if (bOk[e]) copy16_g2l(bG[e], &sB[0][ldsOff[e]]);
    }
    copy_join();
    __syncthreads();

    for (int kt = 0; kt < ktiles; ++kt) {
        const int cur = kt & 1;
        if (kt + 1 < ktiles) {
            const int nb = (kt + 1) & 1;
            const int k0 = (kt + 1) << 5;
#pragma unroll
            for (int e = 0; e < 2; ++e) {
                copy16_g2l(aG[e] + k0, &sA[nb][ldsOff[e]]);
                if (bOk[e]) copy16_g2l(bG[e] + k0, &sB[nb][ldsOff[e]]);
            }
        }

        const _Float16* cA = sA[cur];
        const _Float16* cB = sB[cur];
        v16h a0 = load_a32(cA, wr * 32 + (lane & 15), lane >> 4);
        v16h a1 = load_a32(cA, wr * 32 + 16 + (lane & 15), lane >> 4);
        v16h b[4];
#pragma unroll
        for (int j = 0; j < 4; ++j)
            b[j] = load_b32(cB, wc * 64 + j * 16 + (lane & 15), (lane >> 4) * 16);
#pragma unroll
        for (int j = 0; j < 4; ++j) {
            acc[0][j] = wmma_f16(a0, b[j], acc[0][j]);
            acc[1][j] = wmma_f16(a1, b[j], acc[1][j]);
        }

        copy_join();
        __syncthreads();
    }

    // epilogue: C/D layout VGPR r -> row r + 8*(lane>=16), col lane&15
    const int col = lane & 15;
    const int hs  = (lane >> 4) * 8;
#pragma unroll
    for (int i = 0; i < 2; ++i) {
#pragma unroll
        for (int j = 0; j < 4; ++j) {
            int nb = bn + wc * 64 + j * 16 + col;
            if (nb >= N) continue;
#pragma unroll
            for (int r = 0; r < 8; ++r) {
                int mb = bm + wr * 32 + i * 16 + hs + r;
                float v = acc[i][j][r];
                if (bias) v += bias[nb];
                if (act == 1)      v = 0.5f * v * (1.0f + erff(v * 0.70710678118654752f));
                else if (act == 2) v = tanhf(v);
                if (res) v += res[(size_t)mb * N + nb];
                if (Cf) Cf[(size_t)mb * N + nb] = v;
                else    Ch[(size_t)mb * N + nb] = (_Float16)v;
            }
        }
    }
}

// ---------------------------------------------------------------------------
// Flash attention on qkv16[8192,768] f16 (q|k|v each [N,256]; head h cols h*32..).
// Block = 128 threads (4 waves), grid = 1024. Wave owns 16 query rows.
// 32-key tiles: K staged by async b128 DMA (double-buffered), V transposed
// manually, P bounced through per-wave LDS, online softmax. Output f16.
// ---------------------------------------------------------------------------
__global__ __launch_bounds__(128) void attn_wmma_kernel(
    const _Float16* __restrict__ qkv, _Float16* __restrict__ O)
{
    __shared__ __align__(16) _Float16 Qs[64 * 32];      // [m][d]
    __shared__ __align__(16) _Float16 Ks[2][32 * 32];   // [key][d] (B-layout K^T)
    __shared__ __align__(16) _Float16 Vt[2][32 * 32];   // [d][key] (B-layout V)
    __shared__ __align__(16) _Float16 Ps[4][16 * 32];   // per-wave P [row][key]

    const int t    = threadIdx.x;
    const int lane = t & 31;
    const int w    = t >> 5;
    const int qblk = blockIdx.x & 15;
    const int h    = (blockIdx.x >> 4) & 7;
    const int b    = blockIdx.x >> 7;
    const size_t tokbase = (size_t)b * 1024;
    const int qbase = qblk * 64;
    const float SCALE = 0.17677669529663687f;   // 32^-0.5

    // stage scaled Q tile (64x32): v8h loads, scale, v8h store
#pragma unroll
    for (int e = 0; e < 2; ++e) {
        int i = e * 128 + t;                // 0..255
        int k8 = i & 3, m = i >> 2;         // chunk of 8 d, row m
        v8h hv = *(const v8h*)(qkv + (tokbase + qbase + m) * 768 + h * 32 + k8 * 8);
        v8h sv;
#pragma unroll
        for (int j = 0; j < 8; ++j) sv[j] = (_Float16)((float)hv[j] * SCALE);
        *(v8h*)&Qs[m * 32 + k8 * 8] = sv;
    }

    // K/V staging: 1 chunk of 8 halves per thread per matrix per tile
    const int k8 = t & 3, kj = t >> 2;      // d-chunk, key row 0..31
    const _Float16* kG = qkv + (tokbase + kj) * 768 + 256 + h * 32 + k8 * 8;
    const _Float16* vG = qkv + (tokbase + kj) * 768 + 512 + h * 32 + k8 * 8;
    const int kLds = kj * 32 + k8 * 8;

    // prologue: stage key tile 0
    copy16_g2l(kG, &Ks[0][kLds]);
    {
        v8h hv = *(const v8h*)(vG);
#pragma unroll
        for (int j = 0; j < 8; ++j) Vt[0][(k8 * 8 + j) * 32 + kj] = hv[j];
    }
    copy_join();
    __syncthreads();

    const v16h aq = load_a32(Qs + w * 16 * 32, lane & 15, lane >> 4);
    const v8f zacc = {};
    v8f o0 = {}, o1 = {};
    float rm[8], rs[8];
#pragma unroll
    for (int r = 0; r < 8; ++r) { rm[r] = -1e30f; rs[r] = 0.0f; }

    for (int kt = 0; kt < 32; ++kt) {
        const int cur = kt & 1;
        v8h rv;
        if (kt + 1 < 32) {
            const size_t off = (size_t)(kt + 1) * 32 * 768;
            copy16_g2l(kG + off, &Ks[cur ^ 1][kLds]);   // async into next buffer
            rv = *(const v8h*)(vG + off);
        }

        // S = Q K^T for 32 keys
        v16h bk0 = load_b32(Ks[cur], lane & 15, (lane >> 4) * 16);
        v16h bk1 = load_b32(Ks[cur], 16 + (lane & 15), (lane >> 4) * 16);
        v8f s0 = wmma_f16(aq, bk0, zacc);
        v8f s1 = wmma_f16(aq, bk1, zacc);

        // online softmax (rows live within 16-lane halves)
        float corr[8];
#pragma unroll
        for (int r = 0; r < 8; ++r) {
            float mx = fmaxf(s0[r], s1[r]);
#pragma unroll
            for (int off = 8; off >= 1; off >>= 1)
                mx = fmaxf(mx, __shfl_xor(mx, off, 32));
            float mnew = fmaxf(rm[r], mx);
            float p0 = __expf(s0[r] - mnew);
            float p1 = __expf(s1[r] - mnew);
            float ts = p0 + p1;
#pragma unroll
            for (int off = 8; off >= 1; off >>= 1)
                ts += __shfl_xor(ts, off, 32);
            float c = __expf(rm[r] - mnew);
            rm[r] = mnew;
            rs[r] = rs[r] * c + ts;
            corr[r] = c;
            s0[r] = p0; s1[r] = p1;
        }
#pragma unroll
        for (int r = 0; r < 8; ++r) { o0[r] *= corr[r]; o1[r] *= corr[r]; }

        // P -> per-wave LDS (re-layout acc -> A-fragment)
        {
            _Float16* pw = Ps[w];
            const int prow = (lane >> 4) * 8;
            const int pcol = lane & 15;
#pragma unroll
            for (int r = 0; r < 8; ++r) {
                pw[(prow + r) * 32 + pcol]      = (_Float16)s0[r];
                pw[(prow + r) * 32 + 16 + pcol] = (_Float16)s1[r];
            }
        }
        // V transpose-scatter into next buffer
        if (kt + 1 < 32) {
#pragma unroll
            for (int j = 0; j < 8; ++j)
                Vt[cur ^ 1][(k8 * 8 + j) * 32 + kj] = rv[j];
        }
        __syncthreads();   // P visible; Vt[cur] still intact

        v16h ap  = load_a32(Ps[w], lane & 15, lane >> 4);
        v16h bv0 = load_b32(Vt[cur], lane & 15, (lane >> 4) * 16);
        v16h bv1 = load_b32(Vt[cur], 16 + (lane & 15), (lane >> 4) * 16);
        o0 = wmma_f16(ap, bv0, o0);
        o1 = wmma_f16(ap, bv1, o1);

        copy_join();       // async K copy for kt+1 complete before its readers
        __syncthreads();   // all reads of cur buffers done before overwrite
    }

    // normalize and write O[token][h*32+d] (f16)
    const int col = lane & 15;
    const int hs  = (lane >> 4) * 8;
#pragma unroll
    for (int r = 0; r < 8; ++r) {
        size_t tok = tokbase + qbase + w * 16 + hs + r;
        float inv = 1.0f / rs[r];
        O[tok * 256 + h * 32 + col]      = (_Float16)(o0[r] * inv);
        O[tok * 256 + h * 32 + 16 + col] = (_Float16)(o1[r] * inv);
    }
}

// ---------------------------------------------------------------------------
// LayerNorm over D=512, one block (256 threads) per row.
// Writes f32 (yf) and/or f16 (yh); either may be null. In-place safe for yf.
// ---------------------------------------------------------------------------
__global__ __launch_bounds__(256) void ln_kernel(
    const float* __restrict__ x, const float* __restrict__ g,
    const float* __restrict__ bta, float* __restrict__ yf,
    _Float16* __restrict__ yh)
{
    __shared__ float red0[256];
    __shared__ float red1[256];
    const int t = threadIdx.x;
    const size_t row = blockIdx.x;
    const float* xr = x + row * 512;
    float v0 = xr[t], v1 = xr[t + 256];
    red0[t] = v0 + v1;
    red1[t] = v0 * v0 + v1 * v1;
    __syncthreads();
    for (int off = 128; off >= 1; off >>= 1) {
        if (t < off) { red0[t] += red0[t + off]; red1[t] += red1[t + off]; }
        __syncthreads();
    }
    float mean = red0[0] * (1.0f / 512.0f);
    float var  = red1[0] * (1.0f / 512.0f) - mean * mean;
    float rstd = rsqrtf(var + 1e-5f);
    float r0 = (v0 - mean) * rstd * g[t]       + bta[t];
    float r1 = (v1 - mean) * rstd * g[t + 256] + bta[t + 256];
    if (yf) { yf[row * 512 + t] = r0; yf[row * 512 + t + 256] = r1; }
    if (yh) { yh[row * 512 + t] = (_Float16)r0; yh[row * 512 + t + 256] = (_Float16)r1; }
}

// ---------------------------------------------------------------------------
// patchify: x[8,3,256,256] f32 -> patches[8192,192] f16; v = (p1*8+p2)*3 + c
// ---------------------------------------------------------------------------
__global__ __launch_bounds__(256) void patchify_kernel(
    const float* __restrict__ x, _Float16* __restrict__ p)
{
    int idx = blockIdx.x * 256 + threadIdx.x;
    if (idx >= 8 * 1024 * 192) return;
    int v  = idx % 192;
    int n  = (idx / 192) % 1024;
    int b  = idx / (192 * 1024);
    int c  = v % 3;
    int pp = v / 3;
    int p2 = pp & 7, p1 = pp >> 3;
    int hy = n >> 5, wx = n & 31;
    p[idx] = (_Float16)x[(((size_t)b * 3 + c) * 256 + hy * 8 + p1) * 256 + wx * 8 + p2];
}

__global__ __launch_bounds__(256) void unpatchify_kernel(
    const float* __restrict__ p, float* __restrict__ img)
{
    int idx = blockIdx.x * 256 + threadIdx.x;
    if (idx >= 8 * 3 * 256 * 256) return;
    int xx = idx & 255;
    int y  = (idx >> 8) & 255;
    int c  = (idx / (256 * 256)) % 3;
    int b  = idx / (3 * 256 * 256);
    int hy = y >> 3, p1 = y & 7, wx = xx >> 3, p2 = xx & 7;
    int n = hy * 32 + wx;
    int v = (p1 * 8 + p2) * 3 + c;
    img[idx] = p[((size_t)b * 1024 + n) * 192 + v];
}

// ---------------------------------------------------------------------------
// Host orchestration
// ---------------------------------------------------------------------------
extern "C" void kernel_launch(void* const* d_in, const int* in_sizes, int n_in,
                              void* d_out, int out_size, void* d_ws, size_t ws_size,
                              hipStream_t stream) {
    (void)in_sizes; (void)n_in; (void)out_size; (void)ws_size;
    const float* x     = (const float*)d_in[0];
    const float* pe_w  = (const float*)d_in[1];
    const float* pe_b  = (const float*)d_in[2];
    const float* head1 = (const float*)d_in[25];
    const float* head2 = (const float*)d_in[26];

    const size_t MB1 = 1024 * 1024;
    char* ws = (char*)d_ws;
    float*    tok     = (float*)(ws);                    // 16 MB f32 residual stream
    _Float16* hbuf16  = (_Float16*)(ws + 16 * MB1);      //  8 MB
    _Float16* qkvb16  = (_Float16*)(ws + 24 * MB1);      // 12 MB
    _Float16* attnO16 = (_Float16*)(ws + 36 * MB1);      //  4 MB
    _Float16* ffh16   = (_Float16*)(ws + 40 * MB1);      // 32 MB
    _Float16* patch16 = (_Float16*)(ws + 72 * MB1);      //  3 MB
    float*    out192  = (float*)(ws + 76 * MB1);         //  6 MB
    _Float16* wtpool  = (_Float16*)(ws + 82 * MB1);      // ~45 MB

    const int M = 8192;
    const int GM = M / 128;
    dim3 blk(256);

    // ---- weight pre-pass: transpose+convert all weights to f16 [N][K] ----
    size_t woff = 0;
    auto alloc_wt = [&](size_t n) { _Float16* p = wtpool + woff; woff += n; return p; };
    auto WT = [&](const float* W, _Float16* Wt, int K, int N) {
        int n = K * N;
        wt_kernel<<<dim3((n + 255) / 256), blk, 0, stream>>>(W, Wt, K, N);
    };

    _Float16* wt_pe = alloc_wt(192 * 512);
    WT(pe_w, wt_pe, 192, 512);
    _Float16 *wt_qkv[2], *wt_out[2], *wt_ff1[2], *wt_ff2[2];
    for (int ph = 0; ph < 2; ++ph) {
        int base = 3 + ph * 11;
        const float* qkv_w = (const float*)d_in[base + 2];
        const float* out_w = (const float*)d_in[base + 3];
        const float* ff1_w = (const float*)d_in[base + 7];
        const float* ff2_w = (const float*)d_in[base + 8];
        wt_qkv[ph] = alloc_wt(4 * 512 * 768);
        wt_out[ph] = alloc_wt(4 * 256 * 512);
        wt_ff1[ph] = alloc_wt(4 * 512 * 2048);
        wt_ff2[ph] = alloc_wt(4 * 2048 * 512);
        for (int l = 0; l < 4; ++l) {
            WT(qkv_w + (size_t)l * 512 * 768,  wt_qkv[ph] + (size_t)l * 512 * 768,  512, 768);
            WT(out_w + (size_t)l * 256 * 512,  wt_out[ph] + (size_t)l * 256 * 512,  256, 512);
            WT(ff1_w + (size_t)l * 512 * 2048, wt_ff1[ph] + (size_t)l * 512 * 2048, 512, 2048);
            WT(ff2_w + (size_t)l * 2048 * 512, wt_ff2[ph] + (size_t)l * 2048 * 512, 2048, 512);
        }
    }
    _Float16* wt_h1 = alloc_wt(512 * 2048);  WT(head1, wt_h1, 512, 2048);
    _Float16* wt_h2 = alloc_wt(2048 * 192);  WT(head2, wt_h2, 2048, 192);

    // ---- forward pass ----
    patchify_kernel<<<dim3((8 * 1024 * 192) / 256), blk, 0, stream>>>(x, patch16);
    gemm_wmma_kernel<<<dim3(4, GM), blk, 0, stream>>>(
        patch16, wt_pe, pe_b, nullptr, tok, nullptr, M, 512, 192, 0);

    for (int ph = 0; ph < 2; ++ph) {
        int base = 3 + ph * 11;
        const float* ln1_g = (const float*)d_in[base + 0];
        const float* ln1_b = (const float*)d_in[base + 1];
        const float* out_b = (const float*)d_in[base + 4];
        const float* ln2_g = (const float*)d_in[base + 5];
        const float* ln2_b = (const float*)d_in[base + 6];
        const float* nrm_g = (const float*)d_in[base + 9];
        const float* nrm_b = (const float*)d_in[base + 10];

        for (int l = 0; l < 4; ++l) {
            ln_kernel<<<dim3(M), blk, 0, stream>>>(
                tok, ln1_g + l * 512, ln1_b + l * 512, nullptr, hbuf16);
            gemm_wmma_kernel<<<dim3(6, GM), blk, 0, stream>>>(
                hbuf16, wt_qkv[ph] + (size_t)l * 512 * 768, nullptr, nullptr,
                nullptr, qkvb16, M, 768, 512, 0);
            attn_wmma_kernel<<<dim3(1024), dim3(128), 0, stream>>>(qkvb16, attnO16);
            gemm_wmma_kernel<<<dim3(4, GM), blk, 0, stream>>>(
                attnO16, wt_out[ph] + (size_t)l * 256 * 512, out_b + l * 512, tok,
                tok, nullptr, M, 512, 256, 0);
            ln_kernel<<<dim3(M), blk, 0, stream>>>(
                tok, ln2_g + l * 512, ln2_b + l * 512, nullptr, hbuf16);
            gemm_wmma_kernel<<<dim3(16, GM), blk, 0, stream>>>(
                hbuf16, wt_ff1[ph] + (size_t)l * 512 * 2048, nullptr, nullptr,
                nullptr, ffh16, M, 2048, 512, 1);
            gemm_wmma_kernel<<<dim3(4, GM), blk, 0, stream>>>(
                ffh16, wt_ff2[ph] + (size_t)l * 2048 * 512, nullptr, tok,
                tok, nullptr, M, 512, 2048, 0);
        }
        if (ph == 0) {
            // encoder final norm feeds decoder residual stream (f32, in-place)
            ln_kernel<<<dim3(M), blk, 0, stream>>>(tok, nrm_g, nrm_b, tok, nullptr);
        } else {
            // decoder final norm only feeds head1 (f16)
            ln_kernel<<<dim3(M), blk, 0, stream>>>(tok, nrm_g, nrm_b, nullptr, hbuf16);
        }
    }

    gemm_wmma_kernel<<<dim3(16, GM), blk, 0, stream>>>(
        hbuf16, wt_h1, nullptr, nullptr, nullptr, ffh16, M, 2048, 512, 2);
    gemm_wmma_kernel<<<dim3(2, GM), blk, 0, stream>>>(
        ffh16, wt_h2, nullptr, nullptr, out192, nullptr, M, 192, 2048, 0);
    unpatchify_kernel<<<dim3((8 * 3 * 256 * 256) / 256), blk, 0, stream>>>(
        out192, (float*)d_out);
}
